// FlowAttnLoss_43602507989752
// MI455X (gfx1250) — matrix operand order repaired and checked by reading, not compiled, not run on hardware
//
#include <hip/hip_runtime.h>
#include <hip/hip_fp16.h>
#include <math.h>

#define HW     9216
#define CH     64
#define Wd     96
#define Hd     96
#define BATCH  2
#define EPSF   1e-8f
#define NCHUNK 16
#define MBLOCK 4     // column tiles (of 16) per wave

typedef __attribute__((ext_vector_type(16))) _Float16     v16h;
typedef __attribute__((ext_vector_type(8)))  float        v8f;
typedef __attribute__((ext_vector_type(4)))  unsigned int u4;

union Frag {
    v16h         v;
    u4           q[2];
    unsigned int u[8];
};

// ---- order-preserving float <-> uint key (for atomicMax on floats) ----
__device__ __forceinline__ unsigned int fkey(float f) {
    unsigned int u = __float_as_uint(f);
    return (u & 0x80000000u) ? ~u : (u | 0x80000000u);
}
__device__ __forceinline__ float funkey(unsigned int k) {
    unsigned int u = (k & 0x80000000u) ? (k ^ 0x80000000u) : ~k;
    return __uint_as_float(u);
}

// ---------------- kernel 0: init scratch + output ----------------
__global__ void fal_init_kernel(unsigned int* __restrict__ keys, float* __restrict__ out) {
    int i = blockIdx.x * blockDim.x + threadIdx.x;
    if (i < BATCH * HW) keys[i] = 0u;   // below any real fkey()
    if (i == 0) out[0] = 0.0f;
}

// ---------------- kernel 1: norms + f16 normalized matrices ----------------
// A[b][n][c] = source_norm (GEMM left operand, row-major over positions)
// B[b][m][c] = target_norm (GEMM right operand, per-column contiguous channels)
__global__ void fal_prep_kernel(const float* __restrict__ tgt, const float* __restrict__ src,
                                _Float16* __restrict__ Af, _Float16* __restrict__ Bf,
                                float* __restrict__ tnorm) {
    int i = blockIdx.x * blockDim.x + threadIdx.x;
    if (i >= BATCH * HW) return;
    int b = i / HW, j = i % HW;
    const float* t = tgt + (size_t)b * CH * HW + j;
    const float* s = src + (size_t)b * CH * HW + j;
    float st = 0.0f, ss = 0.0f;
    #pragma unroll 8
    for (int c = 0; c < CH; ++c) {
        float tv = t[(size_t)c * HW]; st += tv * tv;
        float sv = s[(size_t)c * HW]; ss += sv * sv;
    }
    float tn = sqrtf(st), sn = sqrtf(ss);
    tnorm[i] = tn;
    float it = 1.0f / (tn + EPSF);
    float is = 1.0f / (sn + EPSF);
    _Float16* arow = Af + (size_t)i * CH;
    _Float16* brow = Bf + (size_t)i * CH;
    #pragma unroll 8
    for (int c = 0; c < CH; ++c) {
        arow[c] = (_Float16)(s[(size_t)c * HW] * is);
        brow[c] = (_Float16)(t[(size_t)c * HW] * it);
    }
}

// ---------------- kernel 2: nearest-up flow, bilinear warp, cosine vs target ----------------
__global__ void fal_warp_cs_kernel(const float* __restrict__ tgt, const float* __restrict__ src,
                                   const float* __restrict__ flow, const float* __restrict__ tnorm,
                                   float* __restrict__ cs) {
    int i = blockIdx.x * blockDim.x + threadIdx.x;
    if (i >= BATCH * HW) return;
    int b = i / HW, j = i % HW;
    int y = j / Wd, x = j % Wd;

    const float* fl = flow + (size_t)b * 2 * 576;
    int fidx = (y >> 2) * 24 + (x >> 2);                 // nearest 96->24
    float fx = fl[fidx];
    float fy = fl[576 + fidx];

    // px = (base_x+1)*0.5*(W-1) + flow*(W-1)/W  simplifies to:
    float px = (float)x + fx * ((float)(Wd - 1) / (float)Wd);
    float py = (float)y + fy * ((float)(Hd - 1) / (float)Hd);
    float x0f = floorf(px), y0f = floorf(py);
    int x0 = (int)x0f, y0 = (int)y0f;
    float wx1 = px - x0f, wy1 = py - y0f;
    float wx0 = 1.0f - wx1, wy0 = 1.0f - wy1;
    bool vx0 = (x0 >= 0) & (x0 < Wd);
    bool vx1 = (x0 + 1 >= 0) & (x0 + 1 < Wd);
    bool vy0 = (y0 >= 0) & (y0 < Hd);
    bool vy1 = (y0 + 1 >= 0) & (y0 + 1 < Hd);
    int cx0 = min(max(x0, 0), Wd - 1),     cx1 = min(max(x0 + 1, 0), Wd - 1);
    int cy0 = min(max(y0, 0), Hd - 1),     cy1 = min(max(y0 + 1, 0), Hd - 1);
    float w00 = wx0 * wy0 * (float)(vx0 && vy0);
    float w10 = wx1 * wy0 * (float)(vx1 && vy0);
    float w01 = wx0 * wy1 * (float)(vx0 && vy1);
    float w11 = wx1 * wy1 * (float)(vx1 && vy1);
    int i00 = cy0 * Wd + cx0, i10 = cy0 * Wd + cx1;
    int i01 = cy1 * Wd + cx0, i11 = cy1 * Wd + cx1;

    const float* sb = src + (size_t)b * CH * HW;
    const float* tb = tgt + (size_t)b * CH * HW + j;
    float dot = 0.0f, n1sq = 0.0f;
    #pragma unroll 4
    for (int c = 0; c < CH; ++c) {
        const float* base = sb + (size_t)c * HW;
        float v = w00 * base[i00] + w10 * base[i10] + w01 * base[i01] + w11 * base[i11];
        dot  += v * tb[(size_t)c * HW];
        n1sq += v * v;
    }
    float n1 = sqrtf(n1sq);
    float n2 = tnorm[i];
    cs[i] = dot / (fmaxf(n1, EPSF) * fmaxf(n2, EPSF));
}

// ---------------- kernel 3: WMMA GEMM (9216x9216x64) with streaming column-max ----------------
// Each wave: MBLOCK=4 column tiles (64 columns), B fragments resident in VGPRs.
// Per 16-row step: 4 b128 A-loads feed 8 WMMAs (A reused across 4 tiles).
__global__ __launch_bounds__(256) void fal_gemm_max_kernel(const _Float16* __restrict__ A,
                                                           const _Float16* __restrict__ B,
                                                           unsigned int* __restrict__ keys) {
    const int lane  = threadIdx.x & 31;
    const int wave  = threadIdx.x >> 5;
    const int batch = blockIdx.z;
    const int m0    = (blockIdx.x * 8 + wave) * (16 * MBLOCK);  // wave's 64-column strip
    const int laneN = lane & 15;
    const int laneG = lane >> 4;

    const _Float16* Ab = A + (size_t)batch * HW * CH;
    const _Float16* Bb = B + (size_t)batch * HW * CH;

    // B fragments (K=0..31 and K=32..63 per tile), held for the whole n-loop.
    // Layout: lane holds column N=lane%16; lanes 0-15 carry K=0-15, lanes 16-31 K=16-31.
    Frag b0[MBLOCK], b1[MBLOCK];
    #pragma unroll
    for (int t = 0; t < MBLOCK; ++t) {
        const u4* B4 = (const u4*)(Bb + (size_t)(m0 + t * 16 + laneN) * CH);
        b0[t].q[0] = B4[laneG * 2 + 0];
        b0[t].q[1] = B4[laneG * 2 + 1];
        b1[t].q[0] = B4[4 + laneG * 2 + 0];
        b1[t].q[1] = B4[4 + laneG * 2 + 1];
    }

    v8f runmax[MBLOCK];
    #pragma unroll
    for (int t = 0; t < MBLOCK; ++t)
        #pragma unroll
        for (int k = 0; k < 8; ++k) runmax[t][k] = -3.0e38f;
    const v8f czero = {};

    const int nsteps = HW / NCHUNK / 16;                        // 36
    const int n0     = blockIdx.y * (HW / NCHUNK);
    const u4* A4 = (const u4*)(Ab + (size_t)(n0 + laneN) * CH); // row M=lane%16 of current tile

    for (int s = 0; s < nsteps; ++s) {
        __builtin_prefetch((const void*)(A4 + 16 * (CH / 8)), 0, 1); // next 16-row tile
        // A 16x32 f16 layout: VGPR0-7 K = laneG*16 + {0..15}
        Frag a0, a1;
        a0.q[0] = A4[laneG * 2 + 0];
        a0.q[1] = A4[laneG * 2 + 1];
        a1.q[0] = A4[4 + laneG * 2 + 0];
        a1.q[1] = A4[4 + laneG * 2 + 1];

        #pragma unroll
        for (int t = 0; t < MBLOCK; ++t) {
            v8f d = __builtin_amdgcn_wmma_f32_16x16x32_f16(false, a0.v, false, b0[t].v,
                                                           (short)0, czero, false, false);
            d = __builtin_amdgcn_wmma_f32_16x16x32_f16(false, a1.v, false, b1[t].v,
                                                       (short)0, d, false, false);
            #pragma unroll
            for (int k = 0; k < 8; ++k) runmax[t][k] = fmaxf(runmax[t][k], d[k]);
        }

        A4 += 16 * (CH / 8);   // advance 16 rows
    }

    // Fold 8 accumulator rows, then the two lane-halves (M 0-7 vs 8-15) per column.
    #pragma unroll
    for (int t = 0; t < MBLOCK; ++t) {
        float m = runmax[t][0];
        #pragma unroll
        for (int k = 1; k < 8; ++k) m = fmaxf(m, runmax[t][k]);
        m = fmaxf(m, __shfl_xor(m, 16, 32));
        if (laneG == 0) {
            atomicMax(&keys[(size_t)batch * HW + m0 + t * 16 + laneN], fkey(m));
        }
    }
}

// ---------------- kernel 4: loss map + attention-weighted mean ----------------
__global__ void fal_loss_kernel(const float* __restrict__ cs, const unsigned int* __restrict__ keys,
                                const float* __restrict__ att, float* __restrict__ out) {
    int i = blockIdx.x * blockDim.x + threadIdx.x;
    float term = 0.0f;
    if (i < BATCH * HW) {
        int b = i / HW, j = i % HW;
        int y = j / Wd, x = j % Wd;
        float a    = att[(size_t)b * 576 + (y >> 2) * 24 + (x >> 2)];
        float cmax = funkey(keys[i]);
        float lm   = expf(-cs[i] / (cmax + EPSF));
        term = a * (lm - 0.36787944117144233f) * (1.0f / (float)(BATCH * HW));
    }
    #pragma unroll
    for (int off = 16; off > 0; off >>= 1) term += __shfl_down(term, off, 32);
    __shared__ float sbuf[8];
    if ((threadIdx.x & 31) == 0) sbuf[threadIdx.x >> 5] = term;
    __syncthreads();
    if (threadIdx.x < 8) {
        float v = sbuf[threadIdx.x];
        #pragma unroll
        for (int off = 4; off > 0; off >>= 1) v += __shfl_down(v, off, 8);
        if (threadIdx.x == 0) atomicAdd(out, v);
    }
}

extern "C" void kernel_launch(void* const* d_in, const int* in_sizes, int n_in,
                              void* d_out, int out_size, void* d_ws, size_t ws_size,
                              hipStream_t stream) {
    (void)in_sizes; (void)n_in; (void)out_size; (void)ws_size;
    const float* tgt  = (const float*)d_in[0];   // target_vgg [2,64,96,96]
    const float* src  = (const float*)d_in[1];   // source_vgg [2,64,96,96]
    const float* flow = (const float*)d_in[2];   // flow       [2,2,24,24]
    const float* att  = (const float*)d_in[3];   // attention  [2,1,24,24]
    float* out = (float*)d_out;

    char* ws = (char*)d_ws;
    size_t off = 0;
    _Float16* Af = (_Float16*)(ws + off); off += (size_t)BATCH * HW * CH * sizeof(_Float16);
    _Float16* Bf = (_Float16*)(ws + off); off += (size_t)BATCH * HW * CH * sizeof(_Float16);
    float* tnorm = (float*)(ws + off);    off += (size_t)BATCH * HW * sizeof(float);
    float* cs    = (float*)(ws + off);    off += (size_t)BATCH * HW * sizeof(float);
    unsigned int* keys = (unsigned int*)(ws + off);

    const int NPOS   = BATCH * HW;              // 18432
    const int blocks = (NPOS + 255) / 256;      // 72

    fal_init_kernel<<<blocks, 256, 0, stream>>>(keys, out);
    fal_prep_kernel<<<blocks, 256, 0, stream>>>(tgt, src, Af, Bf, tnorm);
    fal_warp_cs_kernel<<<blocks, 256, 0, stream>>>(tgt, src, flow, tnorm, cs);

    dim3 g(HW / (16 * MBLOCK * 8), NCHUNK, BATCH);   // (18, 16, 2) workgroups of 8 waves
    fal_gemm_max_kernel<<<g, 256, 0, stream>>>(Af, Bf, keys);

    fal_loss_kernel<<<blocks, 256, 0, stream>>>(cs, keys, att, out);
}
